// CHPS_model_20976620273883
// MI455X (gfx1250) — matrix operand order repaired and checked by reading, not compiled, not run on hardware
//
#include <hip/hip_runtime.h>
#include <hip/hip_bf16.h>

typedef __attribute__((ext_vector_type(2)))  __bf16 v2bf;
typedef __attribute__((ext_vector_type(4)))  __bf16 v4bf;
typedef __attribute__((ext_vector_type(8)))  __bf16 v8bf;
typedef __attribute__((ext_vector_type(16))) __bf16 v16bf;
typedef __attribute__((ext_vector_type(8)))  float  v8f;

#define B_ROWS 8192
#define F_IN   4096
#define D_DIM  1024
#define P_RAW  1000
#define P_PAD  1024
#define EPS_F  1e-12f

#define SHUF16(lo, hi) __builtin_shufflevector(lo, hi, 0,1,2,3,4,5,6,7,8,9,10,11,12,13,14,15)

// ---------------------------------------------------------------------------
// Kernel 1: normalize prototypes (f32 -> bf16), pad rows [1000,1024) with 0.
// One wave per row. 1024 elems = 32 lanes x 4 chunks of 8.
// ---------------------------------------------------------------------------
__global__ __launch_bounds__(256) void proto_prep_kernel(
    const float* __restrict__ proto, __bf16* __restrict__ protoN) {
  const int wid  = (blockIdx.x * 256 + threadIdx.x) >> 5;   // row id
  const int lane = threadIdx.x & 31;
  if (wid >= P_PAD) return;
  __bf16* dst = protoN + (size_t)wid * D_DIM;

  if (wid >= P_RAW) {                 // zero padding rows
    v8bf z = {};
    #pragma unroll
    for (int c = 0; c < 4; ++c) *(v8bf*)(dst + (lane + 32 * c) * 8) = z;
    return;
  }

  const float* src = proto + (size_t)wid * D_DIM;
  float4 va[4], vb[4];
  float s = 0.f;
  #pragma unroll
  for (int c = 0; c < 4; ++c) {
    const int ch = (lane + 32 * c) * 8;
    va[c] = *(const float4*)(src + ch);
    vb[c] = *(const float4*)(src + ch + 4);
    s += va[c].x * va[c].x + va[c].y * va[c].y + va[c].z * va[c].z + va[c].w * va[c].w;
    s += vb[c].x * vb[c].x + vb[c].y * vb[c].y + vb[c].z * vb[c].z + vb[c].w * vb[c].w;
  }
  #pragma unroll
  for (int m = 16; m >= 1; m >>= 1) s += __shfl_xor(s, m, 32);
  const float scale = 1.f / fmaxf(sqrtf(s), EPS_F);

  #pragma unroll
  for (int c = 0; c < 4; ++c) {
    v8bf o;
    o[0] = (__bf16)(va[c].x * scale); o[1] = (__bf16)(va[c].y * scale);
    o[2] = (__bf16)(va[c].z * scale); o[3] = (__bf16)(va[c].w * scale);
    o[4] = (__bf16)(vb[c].x * scale); o[5] = (__bf16)(vb[c].y * scale);
    o[6] = (__bf16)(vb[c].z * scale); o[7] = (__bf16)(vb[c].w * scale);
    *(v8bf*)(dst + (lane + 32 * c) * 8) = o;
  }
}

// ---------------------------------------------------------------------------
// Kernel 2: emb = x @ W + b   (f32 in, bf16 WMMA, bf16 out to workspace)
// Block = 256 threads (8 waves) -> 128(M) x 64(N) tile; wave -> 32x32.
// Software-pipelined: next K-tile global loads in flight during WMMA.
// ---------------------------------------------------------------------------
__global__ __launch_bounds__(256) void gemm1_kernel(
    const float* __restrict__ x, const float* __restrict__ W,
    const float* __restrict__ bias, __bf16* __restrict__ emb) {
  __shared__ __align__(16) __bf16 As[128 * 32];  // [m][k] bf16
  __shared__ __align__(16) __bf16 Bs[64 * 32];   // [n][k] bf16 (transposed W)

  const int t     = threadIdx.x;
  const int lane  = t & 31, wid = t >> 5;
  const int waveM = wid & 3, waveN = wid >> 2;
  const int lm    = lane & 15, lh = lane >> 4;
  const int mBlock = blockIdx.y * 128;
  const int nBlock = blockIdx.x * 64;

  // Per-thread staging assignments (fixed across K-loop):
  // A: 4 float4 items f = t + 256*j  -> row = f>>3, kc = (f&7)*4
  // B: 1 item: k-pair kp = t&15 (k = 2*kp), column group ng = t>>4 (nc = 4*ng)
  const int bkp = (t & 15) << 1;     // even k within tile
  const int bng = (t >> 4) << 2;     // 4-column group base

  float4 ra[4], rb0, rb1;
  #pragma unroll
  for (int j = 0; j < 4; ++j) {
    const int f = t + 256 * j;
    ra[j] = *(const float4*)(x + (size_t)(mBlock + (f >> 3)) * F_IN + ((f & 7) << 2));
  }
  rb0 = *(const float4*)(W + (size_t)(bkp + 0) * D_DIM + nBlock + bng);
  rb1 = *(const float4*)(W + (size_t)(bkp + 1) * D_DIM + nBlock + bng);

  v8f acc[2][2] = {};

  for (int kt = 0; kt < F_IN; kt += 32) {
    // --- convert staged registers into LDS tiles
    #pragma unroll
    for (int j = 0; j < 4; ++j) {
      const int f = t + 256 * j;
      v4bf w; w[0] = (__bf16)ra[j].x; w[1] = (__bf16)ra[j].y;
              w[2] = (__bf16)ra[j].z; w[3] = (__bf16)ra[j].w;
      *(v4bf*)(As + (f >> 3) * 32 + ((f & 7) << 2)) = w;
    }
    {
      const float b0[4] = {rb0.x, rb0.y, rb0.z, rb0.w};
      const float b1[4] = {rb1.x, rb1.y, rb1.z, rb1.w};
      #pragma unroll
      for (int c = 0; c < 4; ++c) {          // pack (k, k+1) pair -> one dword
        v2bf p; p[0] = (__bf16)b0[c]; p[1] = (__bf16)b1[c];
        *(v2bf*)(Bs + (bng + c) * 32 + bkp) = p;
      }
    }
    __syncthreads();

    // --- kick off next K-tile global loads (clamped, branch-free)
    const int ktn = (kt + 32 < F_IN) ? kt + 32 : kt;
    #pragma unroll
    for (int j = 0; j < 4; ++j) {
      const int f = t + 256 * j;
      ra[j] = *(const float4*)(x + (size_t)(mBlock + (f >> 3)) * F_IN + ktn + ((f & 7) << 2));
    }
    rb0 = *(const float4*)(W + (size_t)(ktn + bkp + 0) * D_DIM + nBlock + bng);
    rb1 = *(const float4*)(W + (size_t)(ktn + bkp + 1) * D_DIM + nBlock + bng);

    // --- fragments + WMMA on current tile (overlaps with loads above)
    v16bf af[2], bf[2];
    #pragma unroll
    for (int ti = 0; ti < 2; ++ti) {        // A frag: lane row = lm, K halves by lh
      const __bf16* pa = As + (waveM * 32 + ti * 16 + lm) * 32 + lh * 8;
      v8bf lo = *(const v8bf*)pa;           // K = 8*lh + 0..7
      v8bf hi = *(const v8bf*)(pa + 16);    // K = 8*lh + 16..23
      af[ti] = SHUF16(lo, hi);
    }
    #pragma unroll
    for (int tj = 0; tj < 2; ++tj) {        // B frag: lane col = lm, K = 8*lh + 0..7, 8*lh + 16..23
      const __bf16* pb = Bs + (waveN * 32 + tj * 16 + lm) * 32 + lh * 8;
      v8bf lo = *(const v8bf*)pb;
      v8bf hi = *(const v8bf*)(pb + 16);
      bf[tj] = SHUF16(lo, hi);
    }
    #pragma unroll
    for (int ti = 0; ti < 2; ++ti)
      #pragma unroll
      for (int tj = 0; tj < 2; ++tj)
        acc[ti][tj] = __builtin_amdgcn_wmma_f32_16x16x32_bf16(
            false, af[ti], false, bf[tj], (short)0, acc[ti][tj], false, false);
    __syncthreads();
  }

  // epilogue: add bias, store bf16 emb. C layout: VGPR i -> row = 8*lh + i, col = lm.
  #pragma unroll
  for (int tj = 0; tj < 2; ++tj) {
    const int n = nBlock + waveN * 32 + tj * 16 + lm;
    const float bv = bias[n];
    #pragma unroll
    for (int ti = 0; ti < 2; ++ti) {
      const int mBase = mBlock + waveM * 32 + ti * 16 + lh * 8;
      #pragma unroll
      for (int i = 0; i < 8; ++i)
        emb[(size_t)(mBase + i) * D_DIM + n] = (__bf16)(acc[ti][tj][i] + bv);
    }
  }
}

// ---------------------------------------------------------------------------
// Kernel 3: in-place L2 row-normalization of emb (bf16). One wave per row.
// ---------------------------------------------------------------------------
__global__ __launch_bounds__(256) void norm_kernel(__bf16* __restrict__ emb) {
  const int wid  = (blockIdx.x * 256 + threadIdx.x) >> 5;
  const int lane = threadIdx.x & 31;
  if (wid >= B_ROWS) return;
  __bf16* row = emb + (size_t)wid * D_DIM;

  v8bf c[4];
  float s = 0.f;
  #pragma unroll
  for (int cc = 0; cc < 4; ++cc) {
    c[cc] = *(const v8bf*)(row + (lane + 32 * cc) * 8);
    #pragma unroll
    for (int e = 0; e < 8; ++e) { const float f = (float)c[cc][e]; s += f * f; }
  }
  #pragma unroll
  for (int m = 16; m >= 1; m >>= 1) s += __shfl_xor(s, m, 32);
  const float scale = 1.f / fmaxf(sqrtf(s), EPS_F);

  #pragma unroll
  for (int cc = 0; cc < 4; ++cc) {
    v8bf o;
    #pragma unroll
    for (int e = 0; e < 8; ++e) o[e] = (__bf16)((float)c[cc][e] * scale);
    *(v8bf*)(row + (lane + 32 * cc) * 8) = o;
  }
}

// ---------------------------------------------------------------------------
// Kernel 4: out = 100 * (emb_n @ protoN^T) - 100. Same tiling as GEMM1; both
// operands already bf16 and already in [M][K] / [N][K] layout -> pure copies.
// Software-pipelined like GEMM1.
// ---------------------------------------------------------------------------
__global__ __launch_bounds__(256) void gemm2_kernel(
    const __bf16* __restrict__ embN, const __bf16* __restrict__ protoN,
    float* __restrict__ out) {
  __shared__ __align__(16) __bf16 As[128 * 32];
  __shared__ __align__(16) __bf16 Bs[64 * 32];

  const int t     = threadIdx.x;
  const int lane  = t & 31, wid = t >> 5;
  const int waveM = wid & 3, waveN = wid >> 2;
  const int lm    = lane & 15, lh = lane >> 4;
  const int mBlock = blockIdx.y * 128;
  const int nBlock = blockIdx.x * 64;

  // A: 2 chunks f = t + 256*j -> row = f>>2, kc = (f&3)*8 ; B: 1 chunk at t.
  v8bf ra[2], rb;
  #pragma unroll
  for (int j = 0; j < 2; ++j) {
    const int f = t + 256 * j;
    ra[j] = *(const v8bf*)(embN + (size_t)(mBlock + (f >> 2)) * D_DIM + ((f & 3) << 3));
  }
  rb = *(const v8bf*)(protoN + (size_t)(nBlock + (t >> 2)) * D_DIM + ((t & 3) << 3));

  v8f acc[2][2] = {};

  for (int kt = 0; kt < D_DIM; kt += 32) {
    #pragma unroll
    for (int j = 0; j < 2; ++j) {
      const int f = t + 256 * j;
      *(v8bf*)(As + (f >> 2) * 32 + ((f & 3) << 3)) = ra[j];
    }
    *(v8bf*)(Bs + (t >> 2) * 32 + ((t & 3) << 3)) = rb;
    __syncthreads();

    const int ktn = (kt + 32 < D_DIM) ? kt + 32 : kt;
    #pragma unroll
    for (int j = 0; j < 2; ++j) {
      const int f = t + 256 * j;
      ra[j] = *(const v8bf*)(embN + (size_t)(mBlock + (f >> 2)) * D_DIM + ktn + ((f & 3) << 3));
    }
    rb = *(const v8bf*)(protoN + (size_t)(nBlock + (t >> 2)) * D_DIM + ktn + ((t & 3) << 3));

    v16bf af[2], bf[2];
    #pragma unroll
    for (int ti = 0; ti < 2; ++ti) {
      const __bf16* pa = As + (waveM * 32 + ti * 16 + lm) * 32 + lh * 8;
      v8bf lo = *(const v8bf*)pa;
      v8bf hi = *(const v8bf*)(pa + 16);
      af[ti] = SHUF16(lo, hi);
    }
    #pragma unroll
    for (int tj = 0; tj < 2; ++tj) {
      const __bf16* pb = Bs + (waveN * 32 + tj * 16 + lm) * 32 + lh * 8;
      v8bf lo = *(const v8bf*)pb;
      v8bf hi = *(const v8bf*)(pb + 16);
      bf[tj] = SHUF16(lo, hi);
    }
    #pragma unroll
    for (int ti = 0; ti < 2; ++ti)
      #pragma unroll
      for (int tj = 0; tj < 2; ++tj)
        acc[ti][tj] = __builtin_amdgcn_wmma_f32_16x16x32_bf16(
            false, af[ti], false, bf[tj], (short)0, acc[ti][tj], false, false);
    __syncthreads();
  }

  // epilogue: dist = (cos - 1)/0.01 = 100*cos - 100; guard n < 1000
  #pragma unroll
  for (int tj = 0; tj < 2; ++tj) {
    const int n = nBlock + waveN * 32 + tj * 16 + lm;
    if (n < P_RAW) {
      #pragma unroll
      for (int ti = 0; ti < 2; ++ti) {
        const int mBase = mBlock + waveM * 32 + ti * 16 + lh * 8;
        #pragma unroll
        for (int i = 0; i < 8; ++i)
          out[(size_t)(mBase + i) * P_RAW + n] = 100.f * acc[ti][tj][i] - 100.f;
      }
    }
  }
}

// ---------------------------------------------------------------------------
extern "C" void kernel_launch(void* const* d_in, const int* in_sizes, int n_in,
                              void* d_out, int out_size, void* d_ws, size_t ws_size,
                              hipStream_t stream) {
  (void)in_sizes; (void)n_in; (void)out_size; (void)ws_size;
  const float* x     = (const float*)d_in[0];   // [8192, 4096]
  const float* W     = (const float*)d_in[1];   // [4096, 1024]
  const float* bias  = (const float*)d_in[2];   // [1024]
  const float* proto = (const float*)d_in[3];   // [1000, 1024]
  float* out = (float*)d_out;                   // [8192, 1000]

  __bf16* emb    = (__bf16*)d_ws;                       // 8192*1024 bf16 (16.8 MB)
  __bf16* protoN = emb + (size_t)B_ROWS * D_DIM;        // 1024*1024 bf16 (2.1 MB)

  proto_prep_kernel<<<P_PAD / 8, 256, 0, stream>>>(proto, protoN);
  gemm1_kernel<<<dim3(D_DIM / 64, B_ROWS / 128), 256, 0, stream>>>(x, W, bias, emb);
  norm_kernel<<<B_ROWS / 8, 256, 0, stream>>>(emb);
  gemm2_kernel<<<dim3(P_PAD / 64, B_ROWS / 128), 256, 0, stream>>>(emb, protoN, out);
}
